// Attention_83528523973361
// MI455X (gfx1250) — matrix operand order repaired and checked by reading, not compiled, not run on hardware
//
#include <hip/hip_runtime.h>
#include <hip/hip_bf16.h>

// Problem constants (match reference)
#define BQ   8
#define SQ   1024
#define DQ   1024
#define NHQ  16
#define NKVQ 8
#define HDQ  64

typedef __attribute__((ext_vector_type(16))) __bf16 v16bf;
typedef __attribute__((ext_vector_type(8)))  float  v8f;

union AFrag { v16bf v; uint4 q[2]; unsigned u[8]; };
union BFrag { v16bf v; uint4 q[2]; };

static __device__ __forceinline__ unsigned short f2bf(float f) {
  unsigned u = __builtin_bit_cast(unsigned, f);
  u += 0x7FFFu + ((u >> 16) & 1u);          // round-to-nearest-even
  return (unsigned short)(u >> 16);
}
static __device__ __forceinline__ float bf2f(unsigned short h) {
  unsigned u = ((unsigned)h) << 16;
  return __builtin_bit_cast(float, u);
}
static __device__ __forceinline__ v8f wmma_bf16(const v16bf& a, const v16bf& b, const v8f& c) {
  // D = A(16x32 bf16) * B(32x16 bf16) + C(16x16 f32)
  return __builtin_amdgcn_wmma_f32_16x16x32_bf16(false, a, false, b, (short)0, c, false, false);
}

// ---------------------------------------------------------------- prep kernels
__global__ void cvt_kernel(const float* __restrict__ in, unsigned short* __restrict__ out, int n) {
  int i = blockIdx.x * blockDim.x + threadIdx.x;
  if (i < n) out[i] = f2bf(in[i]);
}

// W (K x N, f32 row-major) -> WT (N x K, bf16 row-major)
__global__ void tcvt_kernel(const float* __restrict__ W, unsigned short* __restrict__ WT,
                            int K, int N) {
  int i = blockIdx.x * blockDim.x + threadIdx.x;
  if (i >= K * N) return;
  int n = i / K, k = i % K;
  WT[i] = f2bf(W[(size_t)k * N + n]);
}

// In-place RoPE on bf16 [B*S][heads*HD]; thread per (row, head, pair)
__global__ void rope_kernel(unsigned short* __restrict__ X,
                            const float* __restrict__ cosT,
                            const float* __restrict__ sinT,
                            int heads, int total) {
  int i = blockIdx.x * blockDim.x + threadIdx.x;
  if (i >= total) return;
  int pI = i & 31;
  int hh = (i >> 5) % heads;
  int r  = i / (32 * heads);          // b*S + s
  int s  = r & (SQ - 1);
  unsigned short* q = X + (size_t)r * (heads * HDQ) + hh * HDQ + (pI << 1);
  float cv = cosT[s * 32 + pI], sv = sinT[s * 32 + pI];
  float x1 = bf2f(q[0]), x2 = bf2f(q[1]);
  q[0] = f2bf(x1 * cv - x2 * sv);
  q[1] = f2bf(x1 * sv + x2 * cv);
}

// V [B*S][NKV*HD] -> VT [B][NKV][HD][S]
__global__ void vtrans_kernel(const unsigned short* __restrict__ V,
                              unsigned short* __restrict__ VT, int total) {
  int i = blockIdx.x * blockDim.x + threadIdx.x;
  if (i >= total) return;
  int s = i & (SQ - 1);
  int rest = i >> 10;
  int d  = rest & (HDQ - 1);  rest >>= 6;
  int hk = rest & (NKVQ - 1);
  int b  = rest >> 3;
  VT[i] = V[(size_t)(b * SQ + s) * (NKVQ * HDQ) + hk * HDQ + d];
}

// ---------------------------------------------------------------- WMMA GEMM
// C(MxN) = A(MxK bf16 row-major) * BT(NxK bf16 row-major)^T
// Wave tile: 32(M) x 64(N); K compile-time for software pipelining.
template <bool F32OUT, int KD>
__global__ __launch_bounds__(256) void gemm_bf16_kernel(
    const unsigned short* __restrict__ A,
    const unsigned short* __restrict__ BT,
    void* __restrict__ C, int M, int N) {
  const int lane = threadIdx.x & 31;
  const int wid  = threadIdx.x >> 5;
  const int lm = lane & 15, lh = lane >> 4;
  const int ntN = N >> 6;
  int w = blockIdx.x * 8 + wid;
  int nt = w % ntN, mt = w / ntN;
  if (mt >= (M >> 5)) return;
  const int m0 = mt << 5, n0 = nt << 6;

  // A fragment halves are contiguous: 8 bf16 at row + k0 + 8*lh and +16
  const unsigned short* ar0 = A + (size_t)(m0 + lm) * KD + (lh << 3);
  const unsigned short* ar1 = ar0 + (size_t)16 * KD;
  const unsigned short* br[4];
#pragma unroll
  for (int j = 0; j < 4; ++j)
    br[j] = BT + (size_t)(n0 + (j << 4) + lm) * KD + (lh << 3);

  v8f acc[2][4] = {};
#pragma unroll 2
  for (int k0 = 0; k0 < KD; k0 += 32) {
    AFrag a0, a1;
    a0.q[0] = *(const uint4*)(ar0 + k0);
    a0.q[1] = *(const uint4*)(ar0 + k0 + 16);
    a1.q[0] = *(const uint4*)(ar1 + k0);
    a1.q[1] = *(const uint4*)(ar1 + k0 + 16);
#pragma unroll
    for (int j = 0; j < 4; ++j) {
      BFrag b;
      b.q[0] = *(const uint4*)(br[j] + k0);
      b.q[1] = *(const uint4*)(br[j] + k0 + 16);
      acc[0][j] = wmma_bf16(a0.v, b.v, acc[0][j]);
      acc[1][j] = wmma_bf16(a1.v, b.v, acc[1][j]);
    }
  }
#pragma unroll
  for (int i = 0; i < 2; ++i)
#pragma unroll
    for (int j = 0; j < 4; ++j)
#pragma unroll
      for (int v = 0; v < 8; ++v) {
        int r  = m0 + (i << 4) + (lh << 3) + v;
        int cI = n0 + (j << 4) + lm;
        if (F32OUT) ((float*)C)[(size_t)r * N + cI] = acc[i][j][v];
        else ((unsigned short*)C)[(size_t)r * N + cI] = f2bf(acc[i][j][v]);
      }
}

// ---------------------------------------------------------------- flash attention
// One wave per (b, h, 16-row q tile). Block-causal mask at 8-row blocks.
__global__ __launch_bounds__(256) void attn_kernel(
    const unsigned short* __restrict__ Q,    // [B*S][NH*HD] bf16 (RoPE'd)
    const unsigned short* __restrict__ Km,   // [B*S][NKV*HD] bf16 (RoPE'd)
    const unsigned short* __restrict__ VT,   // [B][NKV][HD][S] bf16
    unsigned short* __restrict__ O) {        // [B*S][NH*HD] bf16
  __shared__ unsigned short pl[8][16][32];   // per-wave P transpose tile
  const int lane = threadIdx.x & 31;
  const int wid  = threadIdx.x >> 5;
  const int lm = lane & 15, lh = lane >> 4;

  int t  = blockIdx.x * 8 + wid;             // 0 .. B*NH*(S/16)-1
  int qt = t & 63;
  int h  = (t >> 6) & 15;
  int b  = t >> 10;
  int hk = h >> 1;                           // GQA: rep=2
  int qb = qt << 4;

  // Q A-fragments for full head-dim (d = 0..31 and 32..63), loaded once
  const unsigned short* qr =
      Q + (size_t)(b * SQ + qb + lm) * (NHQ * HDQ) + h * HDQ + (lh << 3);
  AFrag aq0, aq1;
  aq0.q[0] = *(const uint4*)(qr);
  aq0.q[1] = *(const uint4*)(qr + 16);
  aq1.q[0] = *(const uint4*)(qr + 32);
  aq1.q[1] = *(const uint4*)(qr + 48);

  v8f oacc[4] = {};
  float m[8], l[8];
#pragma unroll
  for (int v = 0; v < 8; ++v) { m[v] = -3.0e38f; l[v] = 0.0f; }

  const int nch = (qb + 16 + 31) >> 5;       // kv range [0, qb+16) in 32-chunks
  for (int c = 0; c < nch; ++c) {
    const int kv0 = c << 5;
    const bool full = (kv0 + 32) <= (qb + 8);   // fully unmasked chunk?

    // scores: two 16x16 tiles, B = K^T rows (contiguous K rows per lane)
    const unsigned short* kr0 = Km + (size_t)(b * SQ + kv0 + lm) * (NKVQ * HDQ) + hk * HDQ;
    const unsigned short* kr1 = kr0 + 16 * (NKVQ * HDQ);
    BFrag b00, b01, b10, b11;
    const uint4* p;
    p = (const uint4*)(kr0 + (lh << 3));      b00.q[0] = p[0]; b00.q[1] = p[2];
    p = (const uint4*)(kr0 + 32 + (lh << 3)); b01.q[0] = p[0]; b01.q[1] = p[2];
    p = (const uint4*)(kr1 + (lh << 3));      b10.q[0] = p[0]; b10.q[1] = p[2];
    p = (const uint4*)(kr1 + 32 + (lh << 3)); b11.q[0] = p[0]; b11.q[1] = p[2];
    v8f s0 = {}, s1 = {};
    s0 = wmma_bf16(aq0.v, b00.v, s0);
    s0 = wmma_bf16(aq1.v, b01.v, s0);
    s1 = wmma_bf16(aq0.v, b10.v, s1);
    s1 = wmma_bf16(aq1.v, b11.v, s1);

    float sc0[8], sc1[8];
#pragma unroll
    for (int v = 0; v < 8; ++v) { sc0[v] = s0[v] * 0.125f; sc1[v] = s1[v] * 0.125f; }
    if (!full) {
#pragma unroll
      for (int v = 0; v < 8; ++v) {
        int qblk = (qb + (lh << 3) + v) >> 3;
        if (((kv0 + lm) >> 3)      > qblk) sc0[v] = -3.0e38f;
        if (((kv0 + 16 + lm) >> 3) > qblk) sc1[v] = -3.0e38f;
      }
    }
    // row max across the 16 lanes of this half (row = v + 8*lh per VGPR slot)
    float rm[8];
#pragma unroll
    for (int v = 0; v < 8; ++v) rm[v] = fmaxf(sc0[v], sc1[v]);
    for (int d = 1; d < 16; d <<= 1)
#pragma unroll
      for (int v = 0; v < 8; ++v) rm[v] = fmaxf(rm[v], __shfl_xor(rm[v], d, 32));

    float p0[8], p1[8], rs[8];
#pragma unroll
    for (int v = 0; v < 8; ++v) {
      float mn    = fmaxf(m[v], rm[v]);
      float scale = __expf(m[v] - mn);
      m[v] = mn;
      p0[v] = __expf(sc0[v] - mn);
      p1[v] = __expf(sc1[v] - mn);
      l[v] *= scale;
#pragma unroll
      for (int n = 0; n < 4; ++n) oacc[n][v] *= scale;
      rs[v] = p0[v] + p1[v];
    }
    for (int d = 1; d < 16; d <<= 1)
#pragma unroll
      for (int v = 0; v < 8; ++v) rs[v] += __shfl_xor(rs[v], d, 32);
#pragma unroll
    for (int v = 0; v < 8; ++v) l[v] += rs[v];

    // transpose P (16q x 32kv) through LDS; intra-wave DS ops are in-order
#pragma unroll
    for (int v = 0; v < 8; ++v) {
      pl[wid][(lh << 3) + v][lm]      = f2bf(p0[v]);
      pl[wid][(lh << 3) + v][16 + lm] = f2bf(p1[v]);
    }
    asm volatile("" ::: "memory");
    AFrag ap;
    {
      const uint4* lp = (const uint4*)&pl[wid][lm][lh << 3];
      ap.q[0] = lp[0];    // kv pattern elems [8*lh,  8*lh+7]
      ap.q[1] = lp[2];    // kv pattern elems [16+8*lh, 16+8*lh+7]
    }
    asm volatile("" ::: "memory");

    // out += P(16x32) * V(32x64); B from VT rows (contiguous along kv)
    const unsigned short* vtb =
        VT + (size_t)((b * NKVQ + hk) * HDQ) * SQ + kv0 + (lh << 3);
#pragma unroll
    for (int n = 0; n < 4; ++n) {
      const uint4* vp = (const uint4*)(vtb + (size_t)((n << 4) + lm) * SQ);
      BFrag bv;
      bv.q[0] = vp[0]; bv.q[1] = vp[2];
      oacc[n] = wmma_bf16(ap.v, bv.v, oacc[n]);
    }
  }

  // normalize and store bf16
#pragma unroll
  for (int n = 0; n < 4; ++n)
#pragma unroll
    for (int v = 0; v < 8; ++v) {
      int qg = qb + (lh << 3) + v;
      O[(size_t)(b * SQ + qg) * (NHQ * HDQ) + h * HDQ + (n << 4) + lm] =
          f2bf(oacc[n][v] / l[v]);
    }
}

// ---------------------------------------------------------------- launcher
extern "C" void kernel_launch(void* const* d_in, const int* in_sizes, int n_in,
                              void* d_out, int out_size, void* d_ws, size_t ws_size,
                              hipStream_t stream) {
  (void)in_sizes; (void)n_in; (void)out_size; (void)ws_size;
  const float* x  = (const float*)d_in[0];
  const float* wq = (const float*)d_in[1];
  const float* wk = (const float*)d_in[2];
  const float* wv = (const float*)d_in[3];
  const float* wo = (const float*)d_in[4];
  const float* ct = (const float*)d_in[5];
  const float* st = (const float*)d_in[6];
  // d_in[7] = start_pos (== 0, unused by the reference math)

  const int MS = BQ * SQ;  // 8192 rows
  size_t off = 0;
  auto carve = [&](size_t elems) {
    unsigned short* pp = (unsigned short*)((char*)d_ws + off);
    off += elems * sizeof(unsigned short);
    off = (off + 255) & ~(size_t)255;
    return pp;
  };
  unsigned short* xb  = carve((size_t)MS * DQ);          // 16 MB
  unsigned short* wqT = carve((size_t)DQ * DQ);          //  2 MB  [N][K]
  unsigned short* wkT = carve((size_t)512 * DQ);         //  1 MB
  unsigned short* wvT = carve((size_t)512 * DQ);         //  1 MB
  unsigned short* woT = carve((size_t)DQ * DQ);          //  2 MB
  unsigned short* Qb  = carve((size_t)MS * NHQ * HDQ);   // 16 MB
  unsigned short* Kb  = carve((size_t)MS * NKVQ * HDQ);  //  8 MB
  unsigned short* Vb  = carve((size_t)MS * NKVQ * HDQ);  //  8 MB
  unsigned short* VTb = carve((size_t)BQ * NKVQ * HDQ * SQ); // 8 MB
  unsigned short* Ob  = xb;  // xb is dead after the QKV GEMMs; reuse for attn out

  cvt_kernel<<<(MS * DQ + 255) / 256, 256, 0, stream>>>(x, xb, MS * DQ);
  tcvt_kernel<<<(DQ * DQ + 255) / 256, 256, 0, stream>>>(wq, wqT, DQ, DQ);
  tcvt_kernel<<<(512 * DQ + 255) / 256, 256, 0, stream>>>(wk, wkT, DQ, 512);
  tcvt_kernel<<<(512 * DQ + 255) / 256, 256, 0, stream>>>(wv, wvT, DQ, 512);
  tcvt_kernel<<<(DQ * DQ + 255) / 256, 256, 0, stream>>>(wo, woT, NHQ * HDQ, DQ);

  gemm_bf16_kernel<false, DQ><<<(MS / 32) * (DQ / 64) / 8, 256, 0, stream>>>(xb, wqT, Qb, MS, DQ);
  gemm_bf16_kernel<false, DQ><<<(MS / 32) * (512 / 64) / 8, 256, 0, stream>>>(xb, wkT, Kb, MS, 512);
  gemm_bf16_kernel<false, DQ><<<(MS / 32) * (512 / 64) / 8, 256, 0, stream>>>(xb, wvT, Vb, MS, 512);

  rope_kernel<<<(MS * NHQ * 32 + 255) / 256, 256, 0, stream>>>(Qb, ct, st, NHQ, MS * NHQ * 32);
  rope_kernel<<<(MS * NKVQ * 32 + 255) / 256, 256, 0, stream>>>(Kb, ct, st, NKVQ, MS * NKVQ * 32);
  vtrans_kernel<<<(BQ * NKVQ * HDQ * SQ + 255) / 256, 256, 0, stream>>>(
      Vb, VTb, BQ * NKVQ * HDQ * SQ);

  attn_kernel<<<(BQ * NHQ * (SQ / 16)) / 8, 256, 0, stream>>>(Qb, Kb, VTb, Ob);

  gemm_bf16_kernel<true, DQ><<<(MS / 32) * (DQ / 64) / 8, 256, 0, stream>>>(
      Ob, woT, d_out, MS, DQ);
}
